// spk_vq_vae_resnet_1915555414438
// MI455X (gfx1250) — compile-verified, matrix operand order checked by reading, not run on hardware
//
#include <hip/hip_runtime.h>
#include <hip/hip_bf16.h>

#define ALPHA 0.99f
#define DDIM 128

typedef __attribute__((ext_vector_type(16))) __bf16 v16bf;
typedef __attribute__((ext_vector_type(8)))  float  v8f;

// LDS tile geometry: 16 codebook rows x 128 bf16 (256B) padded to 272B/row to
// rotate bank groups across rows (64 banks x 4B = 256B period otherwise).
#define TILE_ROW_BYTES 256
#define TILE_ROW_PITCH 272
#define TILE_BYTES     (16 * TILE_ROW_PITCH)

// ---- CDNA5 async global->LDS copy (one b128 per lane) ----------------------
__device__ __forceinline__ void async_copy_b128(unsigned ldsOff, const void* gsrc) {
  // GLOBAL_LOAD_ASYNC_TO_LDS_B128: per-lane LDS[vdst] <- MEM[vaddr], ASYNCcnt++
  asm volatile("global_load_async_to_lds_b128 %0, %1, off"
               :: "v"(ldsOff), "v"(gsrc) : "memory");
}
__device__ __forceinline__ void wait_async0() {
  asm volatile("s_wait_asynccnt 0x0" ::: "memory");
}

// Build a 16-bit WMMA fragment from an f32 row in global memory.
// ISA 7.12.2 (16-bit 16x32 layout): lane L holds row M = L%16; half = L/16.
//   vector elems 0..7  <-> K = dBase + half*8 + (0..7)
//   vector elems 8..15 <-> K = dBase + 16 + half*8 + (0..7)
__device__ __forceinline__ v16bf load_frag_f32(const float* __restrict__ row,
                                               int dBase, int half) {
  const float4 a0 = *reinterpret_cast<const float4*>(row + dBase + half * 8);
  const float4 a1 = *reinterpret_cast<const float4*>(row + dBase + half * 8 + 4);
  const float4 b0 = *reinterpret_cast<const float4*>(row + dBase + 16 + half * 8);
  const float4 b1 = *reinterpret_cast<const float4*>(row + dBase + 16 + half * 8 + 4);
  v16bf f;
  f[0]  = (__bf16)a0.x; f[1]  = (__bf16)a0.y; f[2]  = (__bf16)a0.z; f[3]  = (__bf16)a0.w;
  f[4]  = (__bf16)a1.x; f[5]  = (__bf16)a1.y; f[6]  = (__bf16)a1.z; f[7]  = (__bf16)a1.w;
  f[8]  = (__bf16)b0.x; f[9]  = (__bf16)b0.y; f[10] = (__bf16)b0.z; f[11] = (__bf16)b0.w;
  f[12] = (__bf16)b1.x; f[13] = (__bf16)b1.y; f[14] = (__bf16)b1.z; f[15] = (__bf16)b1.w;
  return f;
}

// Same fragment gather but from a bf16 LDS row: two 16B ds_load_b128 reads.
union FragU { v16bf v; uint4 q[2]; };
__device__ __forceinline__ v16bf load_frag_lds(const unsigned char* rowPtr,
                                               int dBase, int half) {
  const int off0 = (dBase + half * 8) * 2;        // bytes; 16B aligned
  FragU f;
  f.q[0] = *reinterpret_cast<const uint4*>(rowPtr + off0);
  f.q[1] = *reinterpret_cast<const uint4*>(rowPtr + off0 + 32);
  return f.v;
}

// ---- phase 0a: W f32 -> bf16 (row-major, rows contiguous: 256B/row) --------
__global__ void vq_wbf_kernel(const float* __restrict__ W,
                              __bf16* __restrict__ Wbf, int total) {
  int base = (blockIdx.x * blockDim.x + threadIdx.x) * 4;
  if (base < total) {
    float4 v = *reinterpret_cast<const float4*>(W + base);
    union { __bf16 h[4]; uint2 u; } o;
    o.h[0] = (__bf16)v.x; o.h[1] = (__bf16)v.y;
    o.h[2] = (__bf16)v.z; o.h[3] = (__bf16)v.w;
    *reinterpret_cast<uint2*>(Wbf + base) = o.u;
  }
}

// ---- phase 0b: w2[k] = ||W_k||^2 -------------------------------------------
__global__ void vq_w2_kernel(const float* __restrict__ W, float* __restrict__ w2, int K) {
  int k = blockIdx.x * blockDim.x + threadIdx.x;
  if (k < K) {
    const float* row = W + (size_t)k * DDIM;
    float s = 0.f;
#pragma unroll 8
    for (int d = 0; d < DDIM; ++d) { float v = row[d]; s += v * v; }
    w2[k] = s;
  }
}

// ---- phase 1: distances via bf16 WMMA + argmin -----------------------------
// 256 threads = 8 waves; each wave owns 16 Z rows (A resident in VGPRs).
// W tiles (16 codes x 128 dims, bf16) are async-DMA'd into double-buffered
// LDS once per block and consumed by all 8 waves.
__global__ __launch_bounds__(256) void vq_dist_argmin_kernel(
    const float* __restrict__ Z, const __bf16* __restrict__ Wbf,
    const float* __restrict__ w2, int* __restrict__ idxOut, int K) {
  __shared__ __align__(16) unsigned char wtile[2][TILE_BYTES];

  const int tid    = threadIdx.x;
  const int lane   = tid & 31;
  const int wave   = tid >> 5;
  const int half   = lane >> 4;     // 0: lanes 0-15, 1: lanes 16-31
  const int l16    = lane & 15;
  const int rowBase = blockIdx.x * 128 + wave * 16;

  // Cooperative async-load mapping: 256 threads x 16B = one 4KB tile.
  const int ldRow = tid >> 4;              // 0..15 (codebook row in tile)
  const int ldCol = (tid & 15) << 4;       // 0..240 step 16 (bytes)
  const unsigned ldsOff0 =
      (unsigned)(size_t)(&wtile[0][ldRow * TILE_ROW_PITCH + ldCol]);
  const unsigned ldsOff1 =
      (unsigned)(size_t)(&wtile[1][ldRow * TILE_ROW_PITCH + ldCol]);
  const unsigned char* gsrc =
      reinterpret_cast<const unsigned char*>(Wbf) + ldRow * TILE_ROW_BYTES + ldCol;

  // A fragments: this wave's 16 Z rows, D=128 as 4 K-slabs of 32 (bf16).
  const float* zrow = Z + (size_t)(rowBase + l16) * DDIM;
  v16bf a0 = load_frag_f32(zrow, 0,  half);
  v16bf a1 = load_frag_f32(zrow, 32, half);
  v16bf a2 = load_frag_f32(zrow, 64, half);
  v16bf a3 = load_frag_f32(zrow, 96, half);

  float best[8];
  int   bidx[8];
#pragma unroll
  for (int r = 0; r < 8; ++r) { best[r] = 3.4e38f; bidx[r] = 0; }

  const int numTiles = K >> 4;            // 128

  // Prime the pipeline with tile 0.
  async_copy_b128(ldsOff0, gsrc);

  for (int t = 0; t < numTiles; ++t) {
    const int p = t & 1;
    wait_async0();       // my wave's slice of tile t has landed in LDS
    __syncthreads();     // all waves' slices landed; all done reading buf[1-p]
    if (t + 1 < numTiles)
      async_copy_b128(p ? ldsOff0 : ldsOff1,
                      gsrc + (size_t)(t + 1) * (16 * TILE_ROW_BYTES));

    const unsigned char* rowPtr = &wtile[p][l16 * TILE_ROW_PITCH];
    v16bf b0 = load_frag_lds(rowPtr, 0,  half);
    v16bf b1 = load_frag_lds(rowPtr, 32, half);
    v16bf b2 = load_frag_lds(rowPtr, 64, half);
    v16bf b3 = load_frag_lds(rowPtr, 96, half);

    v8f c = {};
    c = __builtin_amdgcn_wmma_f32_16x16x32_bf16(false, a0, false, b0, (short)0, c, false, false);
    c = __builtin_amdgcn_wmma_f32_16x16x32_bf16(false, a1, false, b1, (short)0, c, false, false);
    c = __builtin_amdgcn_wmma_f32_16x16x32_bf16(false, a2, false, b2, (short)0, c, false, false);
    c = __builtin_amdgcn_wmma_f32_16x16x32_bf16(false, a3, false, b3, (short)0, c, false, false);

    const int code = (t << 4) + l16;
    const float w2v = w2[code];
    // C layout: VGPR r -> row (rowBase + half*8 + r), col N = l16 -> code.
#pragma unroll
    for (int r = 0; r < 8; ++r) {
      float score = fmaf(-2.0f, c[r], w2v);
      if (score < best[r]) { best[r] = score; bidx[r] = code; }
    }
  }

  // Reduce across the 16 lanes of each half (xor < 16 stays inside the half).
#pragma unroll
  for (int r = 0; r < 8; ++r) {
#pragma unroll
    for (int off = 8; off > 0; off >>= 1) {
      float ob = __shfl_xor(best[r], off, 32);
      int   oi = __shfl_xor(bidx[r], off, 32);
      if (ob < best[r]) { best[r] = ob; bidx[r] = oi; }
    }
  }
  if (l16 == 0) {
#pragma unroll
    for (int r = 0; r < 8; ++r)
      idxOut[rowBase + half * 8 + r] = bidx[r];
  }
}

// ---- phase 2: segment sums via global atomics ------------------------------
__global__ __launch_bounds__(128) void vq_scatter_kernel(
    const float* __restrict__ Z, const int* __restrict__ idx,
    float* __restrict__ sums, float* __restrict__ counts) {
  const int n = blockIdx.x;
  const int d = threadIdx.x;
  const int code = idx[n];
  atomicAdd(&sums[(size_t)code * DDIM + d], Z[(size_t)n * DDIM + d]);
  if (d == 0) atomicAdd(&counts[code], 1.0f);
}

// ---- phase 3: EMA codebook update ------------------------------------------
__global__ __launch_bounds__(128) void vq_update_kernel(
    const float* __restrict__ dictVal, const float* __restrict__ dictCnt,
    const float* __restrict__ sums, const float* __restrict__ counts,
    float* __restrict__ wNew, float* __restrict__ cNew) {
  const int k = blockIdx.x;
  const int d = threadIdx.x;
  const float cnt = counts[k];
  const bool assigned = cnt > 0.0f;
  const float cOld = dictCnt[k];
  const float cN = assigned ? (ALPHA * cOld + (1.0f - ALPHA) * cnt) : cOld;
  const float w = dictVal[(size_t)k * DDIM + d];
  float wN = w;
  if (assigned)
    wN = ALPHA * w + (1.0f - ALPHA) * sums[(size_t)k * DDIM + d] / fmaxf(cN, 1e-12f);
  wNew[(size_t)k * DDIM + d] = wN;
  if (d == 0) cNew[k] = cN;
}

// ---- phase 4: gather quantized vectors + commit loss -----------------------
__global__ __launch_bounds__(128) void vq_gather_loss_kernel(
    const float* __restrict__ Z, const int* __restrict__ idx,
    const float* __restrict__ wNew, float* __restrict__ stOut,
    float* __restrict__ lossAccum, float invN) {
  const int n = blockIdx.x;
  const int d = threadIdx.x;
  const int code = idx[n];
  const float w = wNew[(size_t)code * DDIM + d];
  const float z = Z[(size_t)n * DDIM + d];
  stOut[(size_t)n * DDIM + d] = w;   // straight-through forward = w_new[i]
  float diff = z - w;
  float sq = diff * diff;
#pragma unroll
  for (int off = 16; off > 0; off >>= 1) sq += __shfl_down(sq, off, 32);
  __shared__ float part[4];
  const int wv = threadIdx.x >> 5;
  if ((threadIdx.x & 31) == 0) part[wv] = sq;
  __syncthreads();
  if (threadIdx.x == 0)
    atomicAdd(lossAccum, (part[0] + part[1] + part[2] + part[3]) * invN);
}

extern "C" void kernel_launch(void* const* d_in, const int* in_sizes, int n_in,
                              void* d_out, int out_size, void* d_ws, size_t ws_size,
                              hipStream_t stream) {
  const float* Z       = (const float*)d_in[0];
  const float* dictVal = (const float*)d_in[1];
  const float* dictCnt = (const float*)d_in[2];
  const int D = DDIM;
  const int N = in_sizes[0] / D;   // 65536
  const int K = in_sizes[1] / D;   // 2048

  float* out   = (float*)d_out;
  float* stOut = out;                         // [N*D]
  float* loss  = out + (size_t)N * D;         // [1]
  float* wNew  = loss + 1;                    // [K*D]
  float* cNew  = wNew + (size_t)K * D;        // [K]

  char* ws = (char*)d_ws;
  int*    idx    = (int*)ws;    ws += (size_t)N * sizeof(int);
  float*  counts = (float*)ws;  ws += (size_t)K * sizeof(float);
  float*  sums   = (float*)ws;  ws += (size_t)K * D * sizeof(float);
  float*  w2     = (float*)ws;  ws += (size_t)K * sizeof(float);
  __bf16* Wbf    = (__bf16*)ws;

  hipMemsetAsync(counts, 0, (size_t)K * sizeof(float), stream);
  hipMemsetAsync(sums,   0, (size_t)K * D * sizeof(float), stream);
  hipMemsetAsync(loss,   0, sizeof(float), stream);

  vq_wbf_kernel<<<(K * D / 4 + 255) / 256, 256, 0, stream>>>(dictVal, Wbf, K * D);
  vq_w2_kernel<<<(K + 255) / 256, 256, 0, stream>>>(dictVal, w2, K);
  vq_dist_argmin_kernel<<<N / 128, 256, 0, stream>>>(Z, Wbf, w2, idx, K);
  vq_scatter_kernel<<<N, 128, 0, stream>>>(Z, idx, sums, counts);
  vq_update_kernel<<<K, 128, 0, stream>>>(dictVal, dictCnt, sums, counts, wNew, cNew);
  vq_gather_loss_kernel<<<N, 128, 0, stream>>>(Z, idx, wNew, stOut, loss, 1.0f / (float)N);
}
